// Decoder_76338748719540
// MI455X (gfx1250) — matrix operand order repaired and checked by reading, not compiled, not run on hardware
//
#include <hip/hip_runtime.h>

// ---------------------------------------------------------------------------
// Problem dims
#define V 32000
#define H 512
#define B 16
#define T 128
#define FOURH 2048
#define MROWS 2048          // B*T
#define NUM_RWG 32          // workgroups in recurrence (32 * 16 cols = 512 = H)

typedef __attribute__((ext_vector_type(16))) _Float16 v16h;
typedef __attribute__((ext_vector_type(8)))  _Float16 v8h;
typedef __attribute__((ext_vector_type(8)))  float    v8f;

// ---------------------------------------------------------------------------
// WMMA fragment loaders (contiguous K order k = hb + e, same in A and B, wave32)
//
// A (16x32 f16, row-major source [M][ld]):
//   lane L: m = L&15 ; half-wave K-offset hb = (L&16)?16:0
//   elements 0..7  = A[m][k0+hb+0..7]
//   elements 8..15 = A[m][k0+hb+8..15]
__device__ __forceinline__ v16h load_frag_a(const _Float16* base, int ld,
                                            int m0, int k0, int lane) {
    int m  = m0 + (lane & 15);
    int hb = (lane & 16) ? 16 : 0;
    const v8h* p = (const v8h*)(base + (size_t)m * ld + k0 + hb);
    union { v16h v; v8h h[2]; } u;
    u.h[0] = p[0];      // K = k0+hb    .. +7
    u.h[1] = p[1];      // K = k0+hb+8  .. +15
    return u.v;
}

// B (32x16 f16). Source is weights stored row-major [N][ld] (i.e. B[k][n]=W[n][k]):
//   lane L: n = L&15 ; kb = (L&16)?16:0 ; elements e = W[n][k0+kb+e], e=0..15
__device__ __forceinline__ v16h load_frag_b(const _Float16* w, int ld,
                                            int n0, int k0, int lane) {
    int n  = n0 + (lane & 15);
    int kb = (lane & 16) ? 16 : 0;
    return *(const v16h*)(w + (size_t)n * ld + k0 + kb);
}

// flat shared-aperture address -> wave-relative LDS byte offset (ISA 10.2:
// LDS aperture keeps the LDS offset in addr[31:0])
__device__ __forceinline__ uint32_t lds_offset(const void* p) {
    return (uint32_t)(uintptr_t)p;
}

// per-lane async global->LDS 16-byte copy (ASYNCcnt-tracked)
__device__ __forceinline__ void async_g2l_b128(uint32_t lds_off, const void* gptr) {
    asm volatile("global_load_async_to_lds_b128 %0, %1, off"
                 :: "v"(lds_off), "v"((uint64_t)(uintptr_t)gptr)
                 : "memory");
}

// ---------------------------------------------------------------------------
// Elementwise f32 -> f16 conversion
__global__ __launch_bounds__(256) void f32_to_f16_kernel(const float* __restrict__ src,
                                                         _Float16* __restrict__ dst,
                                                         long n) {
    long i = (long)blockIdx.x * blockDim.x + threadIdx.x;
    long stride = (long)gridDim.x * blockDim.x;
    for (; i < n; i += stride) dst[i] = (_Float16)src[i];
}

// Embedding gather + ReLU -> f16, rows ordered m = b*T + t (teacher-forced shift)
__global__ __launch_bounds__(256) void embed_relu_kernel(const int* __restrict__ tgt,
                                                         const float* __restrict__ emb,
                                                         _Float16* __restrict__ X) {
    int m = blockIdx.x;               // 0..2047
    int t = m % T, b = m / T;
    int tok = (t == 0) ? 1 : tgt[b * T + t - 1];   // BOS = 1
    for (int k = threadIdx.x; k < H; k += blockDim.x) {
        float x = emb[(size_t)tok * H + k];
        X[(size_t)m * H + k] = (_Float16)(x > 0.f ? x : 0.f);
    }
}

// ---------------------------------------------------------------------------
// WMMA GEMM with async-staged B panel:
//   out[m][n] = sum_k A[m][k] * W[n][k] + bias0[n] (+ bias1[n]),  K = 512
// Block = 8 waves, block tile = 16 rows x 128 cols.
// Per K-step the block's 128x32 f16 B panel (8 KB) is DMA'd into LDS with
// global_load_async_to_lds_b128 (2 ops/wave, own ASYNCcnt), double-buffered
// so the copy of tile kt+1 overlaps the WMMA on tile kt.
// grid.x = Ncols/128, grid.y = Mrows/16
__global__ __launch_bounds__(256) void wmma_gemm_async_kernel(
        const _Float16* __restrict__ A,
        const _Float16* __restrict__ Wt,
        const float* __restrict__ bias0,
        const float* __restrict__ bias1,
        float* __restrict__ out,
        int ldout) {
    __shared__ __align__(64) _Float16 bpanel[2][128 * 32];   // 2 x 8 KB

    int tid  = threadIdx.x;
    int lane = tid & 31;
    int wave = tid >> 5;
    int m0   = blockIdx.y * 16;
    int nblk = blockIdx.x * 128;
    int n0   = nblk + wave * 16;

    // issue the 8 KB panel copy for K-tile kt into buffer bufidx:
    // segment s (16 B) -> LDS byte s*16 ; row = s>>2, byte-in-row = (s&3)*16
    auto issue_panel = [&](int bufidx, int kt) {
        const char* src_base = (const char*)(Wt + (size_t)nblk * 512 + kt * 32);
        uint32_t dst_base = lds_offset(&bpanel[bufidx][0]);
#pragma unroll
        for (int j = 0; j < 2; ++j) {
            int s = tid + j * 256;
            const char* src = src_base + (size_t)(s >> 2) * (512 * 2) + (s & 3) * 16;
            async_g2l_b128(dst_base + (uint32_t)s * 16, src);
        }
    };
    // B fragment straight out of the staged LDS panel (row stride 32 halfs)
    auto frag_b_lds = [&](int bufidx) -> v16h {
        int n_local = wave * 16 + (lane & 15);
        int kb = (lane & 16) ? 16 : 0;
        return *(const v16h*)&bpanel[bufidx][n_local * 32 + kb];
    };

    v8f c = {};
    issue_panel(0, 0);
    for (int kt = 0; kt < 16; ++kt) {
        if (kt + 1 < 16) {
            issue_panel((kt + 1) & 1, kt + 1);
            // in-order async completion: <=2 outstanding => tile kt resident
            asm volatile("s_wait_asynccnt 0x2" ::: "memory");
            __builtin_prefetch((const void*)(A + (size_t)(m0 + (lane & 15)) * 512
                                             + (size_t)(kt + 1) * 32), 0, 1);
        } else {
            asm volatile("s_wait_asynccnt 0x0" ::: "memory");
        }
        __syncthreads();                      // panel kt visible to all waves
        v16h a = load_frag_a(A, 512, m0, kt * 32, lane);
        v16h b = frag_b_lds(kt & 1);
        c = __builtin_amdgcn_wmma_f32_16x16x32_f16(false, a, false, b,
                                                   (short)0, c, false, false);
        __syncthreads();                      // protect buffer reuse next iter
    }

    int col = n0 + (lane & 15);
    float bias = bias0[col];
    if (bias1) bias += bias1[col];
    int rb = m0 + ((lane & 16) ? 8 : 0);
#pragma unroll
    for (int r = 0; r < 8; ++r)
        out[(size_t)(rb + r) * ldout + col] = c[r] + bias;
}

// ---------------------------------------------------------------------------
// Sequential LSTM recurrence. 32 WGs, WG w owns h-columns [16w, 16w+16).
// Per step: G = h @ W_hh^T (WMMA, K-split over 8 waves) + Gx, LSTM cell,
// write h2 (f16) into ping-pong h buffer + Hseq, grid barrier.
__global__ __launch_bounds__(256) void lstm_recurrence_kernel(
        const float* __restrict__ Gx,          // [MROWS][4H] f32 (includes biases)
        const _Float16* __restrict__ Whh_h,    // [4H][H] f16
        _Float16* __restrict__ h_pp,           // [2][B][H] f16 ping-pong (buf0 = h0)
        _Float16* __restrict__ Hseq,           // [MROWS][H] f16
        const float* __restrict__ c0,          // [B][H] f32
        float* __restrict__ hf_out,            // [B][H] f32
        float* __restrict__ cf_out,            // [B][H] f32
        unsigned* __restrict__ barrier) {
    __shared__ float red[8][256];
    __shared__ float c_state[256];             // [b][col] for this WG's 16 columns

    int tid  = threadIdx.x;
    int lane = tid & 31;
    int wave = tid >> 5;
    int wg   = blockIdx.x;                     // 0..31
    int ntile = wave & 3;                      // gate block: 0=i 1=f 2=g 3=o
    int khalf = wave >> 2;                     // K-split half
    int gcol_base = ntile * 512 + wg * 16;     // row range in W_hh / col range in Gx

    // init cell state slice
    {
        int r = tid >> 5, L = tid & 31;
        int b = r + ((L & 16) ? 8 : 0);
        int col = L & 15;
        c_state[b * 16 + col] = c0[(size_t)b * H + wg * 16 + col];
    }
    __syncthreads();

    for (int t = 0; t < T; ++t) {
        const _Float16* h_cur = h_pp + (size_t)(t & 1) * B * H;
        _Float16* h_nxt = h_pp + (size_t)((t + 1) & 1) * B * H;

        v8f c = {};
        if (khalf == 0) {   // wave-uniform branch: EXEC stays all-ones for WMMA
            int col = gcol_base + (lane & 15);
            int rb = (lane & 16) ? 8 : 0;
#pragma unroll
            for (int r = 0; r < 8; ++r)
                c[r] = Gx[((size_t)(rb + r) * T + t) * FOURH + col];
        }
#pragma unroll
        for (int i = 0; i < 8; ++i) {
            int k0 = (khalf * 8 + i) * 32;
            v16h a = load_frag_a(h_cur, 512, 0, k0, lane);         // M = 16 batches
            v16h b = load_frag_b(Whh_h, 512, gcol_base, k0, lane);
            c = __builtin_amdgcn_wmma_f32_16x16x32_f16(false, a, false, b,
                                                       (short)0, c, false, false);
        }
#pragma unroll
        for (int r = 0; r < 8; ++r) red[wave][r * 32 + lane] = c[r];
        __syncthreads();

        // LSTM cell: one thread per (batch, col)
        float gi = red[0][tid] + red[4][tid];
        float gf = red[1][tid] + red[5][tid];
        float gg = red[2][tid] + red[6][tid];
        float go = red[3][tid] + red[7][tid];
        int r = tid >> 5, L = tid & 31;
        int b = r + ((L & 16) ? 8 : 0);
        int col = L & 15;
        float cold = c_state[b * 16 + col];
        float i_ = 1.f / (1.f + __expf(-gi));
        float f_ = 1.f / (1.f + __expf(-gf));
        float g_ = tanhf(gg);
        float o_ = 1.f / (1.f + __expf(-go));
        float c2 = f_ * cold + i_ * g_;
        float h2 = o_ * tanhf(c2);
        c_state[b * 16 + col] = c2;

        int hcol = wg * 16 + col;
        _Float16 h2h = (_Float16)h2;
        h_nxt[(size_t)b * H + hcol] = h2h;                         // next-step input
        Hseq[((size_t)b * T + t) * H + hcol] = h2h;                // for vocab GEMM
        if (t == T - 1) {
            hf_out[(size_t)b * H + hcol] = h2;
            cf_out[(size_t)b * H + hcol] = c2;
        }

        // device-wide barrier (monotonic counter, memset to 0 per launch)
        __threadfence();
        __syncthreads();
        if (tid == 0) {
            atomicAdd(barrier, 1u);
            unsigned target = (unsigned)NUM_RWG * (unsigned)(t + 1);
            while (__hip_atomic_load(barrier, __ATOMIC_ACQUIRE,
                                     __HIP_MEMORY_SCOPE_AGENT) < target) {
                __builtin_amdgcn_s_sleep(2);
            }
        }
        __syncthreads();
        __threadfence();
    }
}

// ---------------------------------------------------------------------------
extern "C" void kernel_launch(void* const* d_in, const int* in_sizes, int n_in,
                              void* d_out, int out_size, void* d_ws, size_t ws_size,
                              hipStream_t stream) {
    (void)in_sizes; (void)n_in; (void)out_size; (void)ws_size;
    const float* h0    = (const float*)d_in[1];   // [1,B,H]
    const float* c0    = (const float*)d_in[2];   // [1,B,H]
    const int*   tgt   = (const int*)  d_in[3];   // [B,T]
    const float* emb   = (const float*)d_in[4];   // [V,H]
    const float* W_ih  = (const float*)d_in[5];   // [4H,H]
    const float* W_hh  = (const float*)d_in[6];   // [4H,H]
    const float* b_ih  = (const float*)d_in[7];   // [4H]
    const float* b_hh  = (const float*)d_in[8];   // [4H]
    const float* W_out = (const float*)d_in[9];   // [V,H]
    const float* b_out = (const float*)d_in[10];  // [V]
    float* out = (float*)d_out;

    // workspace carve-up (256B aligned)
    char* ws = (char*)d_ws;
    size_t off = 0;
    auto carve = [&](size_t bytes) -> void* {
        off = (off + 255) & ~(size_t)255;
        void* p = ws + off;
        off += bytes;
        return p;
    };
    _Float16* Wout_h = (_Float16*)carve((size_t)V * H * 2);        // 32.8 MB
    _Float16* Wih_h  = (_Float16*)carve((size_t)FOURH * H * 2);    // 2 MB
    _Float16* Whh_h  = (_Float16*)carve((size_t)FOURH * H * 2);    // 2 MB
    _Float16* X_h    = (_Float16*)carve((size_t)MROWS * H * 2);    // 2 MB
    float*    Gx     = (float*)   carve((size_t)MROWS * FOURH * 4);// 16 MB
    _Float16* Hseq   = (_Float16*)carve((size_t)MROWS * H * 2);    // 2 MB
    _Float16* h_pp   = (_Float16*)carve((size_t)2 * B * H * 2);    // 32 KB
    unsigned* bar    = (unsigned*)carve(256);

    hipMemsetAsync(bar, 0, 256, stream);

    // precision prep: weights/activations -> f16 (f32 accumulate everywhere)
    {
        long n = (long)V * H;
        f32_to_f16_kernel<<<(int)((n + 255) / 256), 256, 0, stream>>>(W_out, Wout_h, n);
    }
    {
        long n = (long)FOURH * H;
        f32_to_f16_kernel<<<(int)((n + 255) / 256), 256, 0, stream>>>(W_ih, Wih_h, n);
        f32_to_f16_kernel<<<(int)((n + 255) / 256), 256, 0, stream>>>(W_hh, Whh_h, n);
    }
    {
        long n = (long)B * H;   // h0 layer 0 -> ping-pong buffer 0
        f32_to_f16_kernel<<<(int)((n + 255) / 256), 256, 0, stream>>>(h0, h_pp, n);
    }
    embed_relu_kernel<<<MROWS, 256, 0, stream>>>(tgt, emb, X_h);

    // Gx = relu(emb[toks]) @ W_ih^T + (b_ih + b_hh)   [2048 x 2048]
    wmma_gemm_async_kernel<<<dim3(FOURH / 128, MROWS / 16), 256, 0, stream>>>(
        X_h, Wih_h, b_ih, b_hh, Gx, FOURH);

    // sequential recurrence -> Hseq (f16), final h/c (f32) into d_out tail
    size_t logitsN = (size_t)B * T * V;
    lstm_recurrence_kernel<<<NUM_RWG, 256, 0, stream>>>(
        Gx, Whh_h, h_pp, Hseq, c0,
        out + logitsN, out + logitsN + (size_t)B * H, bar);

    // logits = Hseq @ W_out^T + b_out   [2048 x 32000]  (store-BW bound)
    wmma_gemm_async_kernel<<<dim3(V / 128, MROWS / 16), 256, 0, stream>>>(
        Hseq, Wout_h, b_out, nullptr, out, V);
}
